// SpatiotemporalAttentionCrossInteractionFuse_DTI_4020089389358
// MI455X (gfx1250) — compile-verified, matrix operand order checked by reading, not run on hardware
//
#include <hip/hip_runtime.h>

// ---------------------------------------------------------------------------
// Problem constants (match reference): B=64, T=1024, V=90, C=512
// Padded node dim Vp=96 (multiple of 32) so all WMMA GEMMs are guard-free.
// ---------------------------------------------------------------------------
constexpr int Bn = 64;
constexpr int Tn = 1024;
constexpr int Vn = 90;
constexpr int Cn = 512;
constexpr int Vp = 96;

constexpr long TV   = (long)Tn * Vn;       // logical input slab per batch
constexpr long BTV  = (long)Bn * TV;
constexpr long CC   = (long)Cn * Cn;
constexpr long CVp  = (long)Cn * Vp;       // (C, Vp) row-major
constexpr long VpC  = (long)Vp * Cn;       // (Vp, C) row-major (transposed ops)
constexpr long VpT  = (long)Vp * Tn;       // (Vp, T) row-major
constexpr long TVp  = (long)Tn * Vp;       // (T, Vp) row-major
constexpr long VpVp = (long)Vp * Vp;

typedef __attribute__((ext_vector_type(16))) __bf16 v16bf;
typedef __attribute__((ext_vector_type(8)))  float  v8f;

__device__ __forceinline__ unsigned short f32_to_bf16(float f) {
  unsigned int bits = __float_as_uint(f);
  bits += 0x7FFFu + ((bits >> 16) & 1u);   // round-to-nearest-even
  return (unsigned short)(bits >> 16);
}

union Frag {
  uint4 q[2];
  v16bf v;
};

// ---------------------------------------------------------------------------
// Batched TN GEMM, bf16 inputs, f32 WMMA accumulate:
//   D[m,n] = sum_k A[m,k] * Bt[n,k]   (both row-major, all dims padded)
// 2x2 register blocking: one wave computes a 32x32 D block (4 accumulators),
// so each K step is 8x global_load_b128 feeding 4x v_wmma_f32_16x16x32_bf16
// (2x the FLOP/byte of unblocked). 4 waves/block stacked along M (128 rows).
// Fragment layout per CDNA5 ISA 7.12.2: element i <-> k = k0 + 8*(lane>=16)
// + i + 8*(i>=8) -> two contiguous 16B chunks per lane, no guards.
// Outputs (compile-time selected): f32 row-major, bf16 row-major, bf16
// transposed (n-major). Values outside (Mlog, Nlog) are stored as 0 so padded
// regions of downstream K dimensions stay clean.
// M and N must be multiples of 32; K a multiple of 32.
// ---------------------------------------------------------------------------
constexpr int GEMM_WAVES = 4;

template <bool F32OUT, bool BFOUT, bool BFTOUT>
__global__ void wmma_gemm_tn(const unsigned short* __restrict__ A,
                             const unsigned short* __restrict__ Bt,
                             float* __restrict__ Df32,
                             unsigned short* __restrict__ Dbf,
                             unsigned short* __restrict__ DbfT,
                             const float* __restrict__ bias,
                             int M, int N, int K, int Mlog, int Nlog,
                             int lda, int ldb, int ldd, int ldt,
                             long sA, long sB, long sD, long sDT) {
  const int wave = threadIdx.x >> 5;
  const int lane = threadIdx.x & 31;
  const int g    = lane >> 4;
  const int r    = lane & 15;
  const int n0   = blockIdx.x * 32;
  const int m0   = (blockIdx.y * GEMM_WAVES + wave) * 32;
  const int b    = blockIdx.z;
  if (m0 >= M) return;                       // wave-uniform; EXEC stays all-1s

  const unsigned short* arow0 = A  + (size_t)b * sA + (size_t)(m0 + r) * lda + 8 * g;
  const unsigned short* arow1 = arow0 + (size_t)16 * lda;
  const unsigned short* brow0 = Bt + (size_t)b * sB + (size_t)(n0 + r) * ldb + 8 * g;
  const unsigned short* brow1 = brow0 + (size_t)16 * ldb;

  v8f acc[2][2] = {};
  for (int k0 = 0; k0 < K; k0 += 32) {
    Frag fa0, fa1, fb0, fb1;
    fa0.q[0] = *(const uint4*)(arow0 + k0);
    fa0.q[1] = *(const uint4*)(arow0 + k0 + 16);
    fa1.q[0] = *(const uint4*)(arow1 + k0);
    fa1.q[1] = *(const uint4*)(arow1 + k0 + 16);
    fb0.q[0] = *(const uint4*)(brow0 + k0);
    fb0.q[1] = *(const uint4*)(brow0 + k0 + 16);
    fb1.q[0] = *(const uint4*)(brow1 + k0);
    fb1.q[1] = *(const uint4*)(brow1 + k0 + 16);
    __builtin_prefetch(arow0 + k0 + 256, 0, 1);   // global_prefetch_b8
    __builtin_prefetch(arow1 + k0 + 256, 0, 1);
    __builtin_prefetch(brow0 + k0 + 256, 0, 1);
    __builtin_prefetch(brow1 + k0 + 256, 0, 1);
    acc[0][0] = __builtin_amdgcn_wmma_f32_16x16x32_bf16(false, fa0.v, false, fb0.v,
                                                        (short)0, acc[0][0], false, false);
    acc[0][1] = __builtin_amdgcn_wmma_f32_16x16x32_bf16(false, fa0.v, false, fb1.v,
                                                        (short)0, acc[0][1], false, false);
    acc[1][0] = __builtin_amdgcn_wmma_f32_16x16x32_bf16(false, fa1.v, false, fb0.v,
                                                        (short)0, acc[1][0], false, false);
    acc[1][1] = __builtin_amdgcn_wmma_f32_16x16x32_bf16(false, fa1.v, false, fb1.v,
                                                        (short)0, acc[1][1], false, false);
  }

#pragma unroll
  for (int tm = 0; tm < 2; ++tm) {
#pragma unroll
    for (int tn = 0; tn < 2; ++tn) {
#pragma unroll
      for (int rr = 0; rr < 8; ++rr) {
        const int m = m0 + tm * 16 + rr + 8 * g;
        const int n = n0 + tn * 16 + r;
        const float bb = bias ? bias[m] : 0.0f;
        const float v = ((m < Mlog) && (n < Nlog)) ? (acc[tm][tn][rr] + bb) : 0.0f;
        if constexpr (F32OUT) Df32[(size_t)b * sD + (size_t)m * ldd + n] = v;
        if constexpr (BFOUT)  Dbf [(size_t)b * sD + (size_t)m * ldd + n] = f32_to_bf16(v);
        if constexpr (BFTOUT) DbfT[(size_t)b * sDT + (size_t)n * ldt + m] = f32_to_bf16(v);
      }
    }
  }
}

// ---------------------------------------------------------------------------
// BatchNorm stats over (B, V) per T channel; ld = row pitch of the tensor.
// ---------------------------------------------------------------------------
__global__ void bn_stats_kernel(const float* __restrict__ x, int ld,
                                float* __restrict__ mu, float* __restrict__ rsig) {
  const int t = blockIdx.x, tid = threadIdx.x;
  __shared__ float s1[256], s2[256];
  const int NBV = Bn * Vn;
  float a = 0.f, q = 0.f;
  for (int i = tid; i < NBV; i += 256) {
    const int b = i / Vn, v = i - b * Vn;
    const float val = x[((size_t)b * Tn + t) * ld + v];
    a += val; q += val * val;
  }
  s1[tid] = a; s2[tid] = q; __syncthreads();
  for (int s = 128; s > 0; s >>= 1) {
    if (tid < s) { s1[tid] += s1[tid + s]; s2[tid] += s2[tid + s]; }
    __syncthreads();
  }
  if (tid == 0) {
    const float m = s1[0] / NBV;
    const float var = s2[0] / NBV - m * m;
    mu[t] = m; rsig[t] = rsqrtf(var + 1e-5f);
  }
}

// x (B,T,V) -> xhatT (B, Vp, T) bf16, BN-normalized, pad rows zeroed.
__global__ void xhatT_bf16_kernel(const float* __restrict__ x,
                                  const float* __restrict__ mu,
                                  const float* __restrict__ rs,
                                  unsigned short* __restrict__ out) {
  const long i = (long)blockIdx.x * blockDim.x + threadIdx.x;
  if (i >= (long)Bn * VpT) return;
  const int t = (int)(i % Tn);
  const long rw = i / Tn;
  const int v = (int)(rw % Vp);
  const int b = (int)(rw / Vp);
  float val = 0.f;
  if (v < Vn) val = (x[(size_t)b * TV + (size_t)t * Vn + v] - mu[t]) * rs[t];
  out[i] = f32_to_bf16(val);
}

// Fold BN gamma/beta into 1x1-conv weight: W'[c,t]=W[c,t]*g[t]; b'[c]=b[c]+W[c,:]@beta
__global__ void fold_weight_kernel(const float* __restrict__ W,
                                   const float* __restrict__ bin,
                                   const float* __restrict__ gamma,
                                   const float* __restrict__ beta,
                                   unsigned short* __restrict__ Wbf,
                                   float* __restrict__ bout) {
  const int c = blockIdx.x, tid = threadIdx.x;
  __shared__ float sm[256];
  float acc = 0.f;
  for (int t = tid; t < Tn; t += 256) {
    const float w = W[(size_t)c * Tn + t];
    Wbf[(size_t)c * Tn + t] = f32_to_bf16(w * gamma[t]);
    acc += w * beta[t];
  }
  sm[tid] = acc; __syncthreads();
  for (int s = 128; s > 0; s >>= 1) { if (tid < s) sm[tid] += sm[tid + s]; __syncthreads(); }
  if (tid == 0) bout[c] = bin[c] + sm[0];
}

__global__ void cvt_bf16_kernel(const float* __restrict__ in,
                                unsigned short* __restrict__ out, long n) {
  const long i = (long)blockIdx.x * blockDim.x + threadIdx.x;
  if (i < n) out[i] = f32_to_bf16(in[i]);
}

// Softmax of one logical row r (trans=0) or column r (trans=1) of an f32
// matrix (pitch ldi); bf16 output written as row r (transOut=0) or column r
// (transOut=1) with pitch ldo.
__global__ void softmax_bf16_kernel(const float* __restrict__ in,
                                    unsigned short* __restrict__ out,
                                    int len, int ldi, int ldo,
                                    int trans, int transOut,
                                    long sIn, long sOut) {
  const int r = blockIdx.x, b = blockIdx.y, tid = threadIdx.x;
  const float* ib = in + (size_t)b * sIn;
  unsigned short* ob = out + (size_t)b * sOut;
  __shared__ float sm[256];
  float lmax = -3.0e38f;
  for (int j = tid; j < len; j += 256) {
    const float v = trans ? ib[(size_t)j * ldi + r] : ib[(size_t)r * ldi + j];
    lmax = fmaxf(lmax, v);
  }
  sm[tid] = lmax; __syncthreads();
  for (int s = 128; s > 0; s >>= 1) { if (tid < s) sm[tid] = fmaxf(sm[tid], sm[tid + s]); __syncthreads(); }
  const float mx = sm[0]; __syncthreads();
  float lsum = 0.f;
  for (int j = tid; j < len; j += 256) {
    const float v = trans ? ib[(size_t)j * ldi + r] : ib[(size_t)r * ldi + j];
    lsum += expf(v - mx);
  }
  sm[tid] = lsum; __syncthreads();
  for (int s = 128; s > 0; s >>= 1) { if (tid < s) sm[tid] += sm[tid + s]; __syncthreads(); }
  const float inv = 1.0f / sm[0];
  for (int j = tid; j < len; j += 256) {
    const float v = trans ? ib[(size_t)j * ldi + r] : ib[(size_t)r * ldi + j];
    const unsigned short o = f32_to_bf16(expf(v - mx) * inv);
    if (transOut) ob[(size_t)j * ldo + r] = o;
    else          ob[(size_t)r * ldo + j] = o;
  }
}

// e_s1m = e_s1 * (sigmoid(e_s1)*SG + (1-sigmoid(e_s1))*I); padded pitch Vp.
__global__ void es_modulate_kernel(const float* __restrict__ es,
                                   const float* __restrict__ sg,
                                   float* __restrict__ out) {
  const long i = (long)blockIdx.x * blockDim.x + threadIdx.x;
  if (i >= (long)Bn * Vn * Vn) return;
  const int col = (int)(i % Vn);
  const long rw = i / Vn;
  const int row = (int)(rw % Vn);
  const int b = (int)(rw / Vn);
  const size_t idx = (size_t)b * VpVp + (size_t)row * Vp + col;
  const float e = es[idx];
  const float a = 1.0f / (1.0f + expf(-e));
  const float id = (row == col) ? 1.0f : 0.0f;
  out[idx] = e * (a * sg[i] + (1.0f - a) * id);
}

__global__ void mean_last_kernel(const float* __restrict__ x, float* __restrict__ out) {
  const int i = blockIdx.x * blockDim.x + threadIdx.x;   // over B*T
  if (i >= Bn * Tn) return;
  const float* p = x + (size_t)i * Vn;
  float s = 0.f;
  for (int v = 0; v < Vn; ++v) s += p[v];
  out[i] = s * (1.0f / Vn);
}

__global__ void mean_mid_kernel(const float* __restrict__ x, float* __restrict__ out) {
  const int i = blockIdx.x * blockDim.x + threadIdx.x;   // over B*V
  if (i >= Bn * Vn) return;
  const int b = i / Vn, v = i - b * Vn;
  const float* p = x + (size_t)b * TV + v;
  float s = 0.f;
  for (int t = 0; t < Tn; ++t) s += p[(size_t)t * Vn];
  out[i] = s * (1.0f / Tn);
}

__global__ void mean_all_kernel(const float* __restrict__ x, float* __restrict__ out) {
  const int b = blockIdx.x, tid = threadIdx.x;
  __shared__ float sm[256];
  float s = 0.f;
  for (long i = tid; i < TV; i += 256) s += x[(size_t)b * TV + i];
  sm[tid] = s; __syncthreads();
  for (int st = 128; st > 0; st >>= 1) { if (tid < st) sm[tid] += sm[tid + st]; __syncthreads(); }
  if (tid == 0) out[b] = sm[0] / (float)TV;
}

// mean over (t,v) of BN-normalized conv output (pitch Vp)
__global__ void proj_mean_kernel(const float* __restrict__ conv,
                                 const float* __restrict__ mu, const float* __restrict__ rs,
                                 const float* __restrict__ gam, const float* __restrict__ bet,
                                 float* __restrict__ out) {
  const int b = blockIdx.x, tid = threadIdx.x;
  __shared__ float sm[256];
  float s = 0.f;
  for (long i = tid; i < TV; i += 256) {
    const int t = (int)(i / Vn);
    const int v = (int)(i % Vn);
    s += (conv[((size_t)b * Tn + t) * Vp + v] - mu[t]) * rs[t] * gam[t] + bet[t];
  }
  sm[tid] = s; __syncthreads();
  for (int st = 128; st > 0; st >>= 1) { if (tid < st) sm[tid] += sm[tid + st]; __syncthreads(); }
  if (tid == 0) out[b] = sm[0] / (float)TV;
}

// fused = w0*x + w1*cross + w2*BN(conv); w = softmax([m_self, m_cross, m_proj])
__global__ void fuse_kernel(const float* __restrict__ x, const float* __restrict__ cross,
                            int crossPerV,
                            const float* __restrict__ conv, const float* __restrict__ mu,
                            const float* __restrict__ rs, const float* __restrict__ gam,
                            const float* __restrict__ bet,
                            const float* __restrict__ m_self, const float* __restrict__ m_cross,
                            const float* __restrict__ m_proj, float* __restrict__ out) {
  const long i = (long)blockIdx.x * blockDim.x + threadIdx.x;
  if (i >= BTV) return;
  const int v = (int)(i % Vn);
  const long rw = i / Vn;
  const int t = (int)(rw % Tn);
  const int b = (int)(rw / Tn);
  const float w0 = m_self[b], w1 = m_cross[b], w2 = m_proj[b];
  const float mx = fmaxf(w0, fmaxf(w1, w2));
  const float e0 = expf(w0 - mx), e1 = expf(w1 - mx), e2 = expf(w2 - mx);
  const float inv = 1.0f / (e0 + e1 + e2);
  const float proj = (conv[((size_t)b * Tn + t) * Vp + v] - mu[t]) * rs[t] * gam[t] + bet[t];
  const float cr = crossPerV ? cross[(size_t)b * Vn + v] : cross[(size_t)b * Tn + t];
  out[i] = (e0 * x[i] + e1 * cr + e2 * proj) * inv;
}

// ---------------------------------------------------------------------------
extern "C" void kernel_launch(void* const* d_in, const int* in_sizes, int n_in,
                              void* d_out, int out_size, void* d_ws, size_t ws_size,
                              hipStream_t stream) {
  (void)in_sizes; (void)n_in; (void)out_size; (void)ws_size;

  const float* x1   = (const float*)d_in[0];
  const float* x2   = (const float*)d_in[1];
  const float* sg   = (const float*)d_in[2];
  const float* g_bn_g  = (const float*)d_in[3];
  const float* g_bn_b  = (const float*)d_in[4];
  const float* g_w     = (const float*)d_in[5];
  const float* g_b     = (const float*)d_in[6];
  const float* th_bn_g = (const float*)d_in[7];
  const float* th_bn_b = (const float*)d_in[8];
  const float* th_w    = (const float*)d_in[9];
  const float* th_b    = (const float*)d_in[10];
  const float* ph_bn_g = (const float*)d_in[11];
  const float* ph_bn_b = (const float*)d_in[12];
  const float* ph_w    = (const float*)d_in[13];
  const float* ph_b    = (const float*)d_in[14];
  const float* W_w     = (const float*)d_in[15];
  const float* W_b     = (const float*)d_in[16];
  const float* W_bn_g  = (const float*)d_in[17];
  const float* W_bn_b  = (const float*)d_in[18];
  // d_in[19..26] (qt/ks/qs/kt) cannot affect outputs: softmax over a (B,1,1)
  // scalar is identically 1, so the cross terms reduce to x means.

  float* out = (float*)d_out;

  // ----- carve workspace (256B-aligned slabs) -----
  char* ws = (char*)d_ws;
  size_t off = 0;
  auto carve = [&](size_t bytes) -> void* {
    void* p = ws + off;
    off = (off + bytes + 255) & ~(size_t)255;
    return p;
  };
  float* mu1  = (float*)carve(Tn * 4);  float* rs1  = (float*)carve(Tn * 4);
  float* mu2  = (float*)carve(Tn * 4);  float* rs2  = (float*)carve(Tn * 4);
  float* muc1 = (float*)carve(Tn * 4);  float* rsc1 = (float*)carve(Tn * 4);
  float* muc2 = (float*)carve(Tn * 4);  float* rsc2 = (float*)carve(Tn * 4);
  float* bg   = (float*)carve(Cn * 4);
  float* bth  = (float*)carve(Cn * 4);
  float* bph  = (float*)carve(Cn * 4);
  float* e_t1   = (float*)carve((size_t)Bn * CC * 4);
  float* e_s1   = (float*)carve((size_t)Bn * VpVp * 4);
  float* e_s1m  = (float*)carve((size_t)Bn * VpVp * 4);
  float* conv1f = (float*)carve((size_t)Bn * TVp * 4);
  float* conv2f = (float*)carve((size_t)Bn * TVp * 4);
  float* x1_tm  = (float*)carve((size_t)Bn * Tn * 4);
  float* x2_sm  = (float*)carve((size_t)Bn * Vn * 4);
  float* meanA1 = (float*)carve(Bn * 4);
  float* meanA2 = (float*)carve(Bn * 4);
  float* p1     = (float*)carve(Bn * 4);
  float* p2     = (float*)carve(Bn * 4);
  unsigned short* x1hatT = (unsigned short*)carve((size_t)Bn * VpT * 2);
  unsigned short* x2hatT = (unsigned short*)carve((size_t)Bn * VpT * 2);
  unsigned short* wg_bf  = (unsigned short*)carve((size_t)Cn * Tn * 2);
  unsigned short* wth_bf = (unsigned short*)carve((size_t)Cn * Tn * 2);
  unsigned short* wph_bf = (unsigned short*)carve((size_t)Cn * Tn * 2);
  unsigned short* Ww_bf  = (unsigned short*)carve((size_t)Tn * Cn * 2);
  unsigned short* g1T  = (unsigned short*)carve((size_t)Bn * VpC * 2);
  unsigned short* g2T  = (unsigned short*)carve((size_t)Bn * VpC * 2);
  unsigned short* th1  = (unsigned short*)carve((size_t)Bn * CVp * 2);
  unsigned short* th1T = (unsigned short*)carve((size_t)Bn * VpC * 2);
  unsigned short* ph1  = (unsigned short*)carve((size_t)Bn * CVp * 2);
  unsigned short* ph1T = (unsigned short*)carve((size_t)Bn * VpC * 2);
  unsigned short* at1  = (unsigned short*)carve((size_t)Bn * CC * 2);
  unsigned short* at2  = (unsigned short*)carve((size_t)Bn * CC * 2);
  unsigned short* as1T = (unsigned short*)carve((size_t)Bn * VpVp * 2);
  unsigned short* as2T = (unsigned short*)carve((size_t)Bn * VpVp * 2);
  unsigned short* tmp1 = (unsigned short*)carve((size_t)Bn * CVp * 2);
  unsigned short* tmp2 = (unsigned short*)carve((size_t)Bn * CVp * 2);
  unsigned short* z1T  = (unsigned short*)carve((size_t)Bn * VpC * 2);
  unsigned short* z2T  = (unsigned short*)carve((size_t)Bn * VpC * 2);

  const int EW = 256;
  auto cdiv = [](long a, long b) { return (int)((a + b - 1) / b); };
  const dim3 gblk(32 * GEMM_WAVES);
  float* NOF = nullptr;

  // ----- BN stats + normalized, transposed bf16 inputs (pads zeroed) -----
  bn_stats_kernel<<<Tn, 256, 0, stream>>>(x1, Vn, mu1, rs1);
  bn_stats_kernel<<<Tn, 256, 0, stream>>>(x2, Vn, mu2, rs2);
  xhatT_bf16_kernel<<<cdiv((long)Bn * VpT, EW), EW, 0, stream>>>(x1, mu1, rs1, x1hatT);
  xhatT_bf16_kernel<<<cdiv((long)Bn * VpT, EW), EW, 0, stream>>>(x2, mu2, rs2, x2hatT);

  // ----- fold BN gamma/beta into conv weights; convert W_w -----
  fold_weight_kernel<<<Cn, 256, 0, stream>>>(g_w,  g_b,  g_bn_g,  g_bn_b,  wg_bf,  bg);
  fold_weight_kernel<<<Cn, 256, 0, stream>>>(th_w, th_b, th_bn_g, th_bn_b, wth_bf, bth);
  fold_weight_kernel<<<Cn, 256, 0, stream>>>(ph_w, ph_b, ph_bn_g, ph_bn_b, wph_bf, bph);
  cvt_bf16_kernel<<<cdiv((long)Tn * Cn, EW), EW, 0, stream>>>(W_w, Ww_bf, (long)Tn * Cn);

  // ----- projections: D = W'(C,T) @ xhatT^T, K=T -----
  {
    dim3 grid(Vp / 32, Cn / (32 * GEMM_WAVES), Bn);
    // g1T/g2T: transposed-only (consumed as Bt in z-chain)
    wmma_gemm_tn<false, false, true><<<grid, gblk, 0, stream>>>(
        wg_bf, x1hatT, NOF, nullptr, g1T, bg, Cn, Vp, Tn, Cn, Vn,
        Tn, Tn, 0, Cn, 0, VpT, 0, VpC);
    wmma_gemm_tn<false, false, true><<<grid, gblk, 0, stream>>>(
        wg_bf, x2hatT, NOF, nullptr, g2T, bg, Cn, Vp, Tn, Cn, Vn,
        Tn, Tn, 0, Cn, 0, VpT, 0, VpC);
    // th1/ph1: both layouts (A of e_t1 / A,Bt of e_s1)
    wmma_gemm_tn<false, true, true><<<grid, gblk, 0, stream>>>(
        wth_bf, x1hatT, NOF, th1, th1T, bth, Cn, Vp, Tn, Cn, Vn,
        Tn, Tn, Vp, Cn, 0, VpT, CVp, VpC);
    wmma_gemm_tn<false, true, true><<<grid, gblk, 0, stream>>>(
        wph_bf, x2hatT, NOF, ph1, ph1T, bph, Cn, Vp, Tn, Cn, Vn,
        Tn, Tn, Vp, Cn, 0, VpT, CVp, VpC);
  }

  // ----- e_t1 = th1 @ ph1^T (K=Vp, pads zero) ; e_s1 = th1^T @ ph1 -----
  {
    dim3 grid(Cn / 32, Cn / (32 * GEMM_WAVES), Bn);
    wmma_gemm_tn<true, false, false><<<grid, gblk, 0, stream>>>(
        th1, ph1, e_t1, nullptr, nullptr, nullptr, Cn, Cn, Vp, Cn, Cn,
        Vp, Vp, Cn, 0, CVp, CVp, CC, 0);
  }
  {
    dim3 grid(Vp / 32, cdiv(Vp, 32 * GEMM_WAVES), Bn);
    wmma_gemm_tn<true, false, false><<<grid, gblk, 0, stream>>>(
        th1T, ph1T, e_s1, nullptr, nullptr, nullptr, Vp, Vp, Cn, Vn, Vn,
        Cn, Cn, Vp, 0, VpC, VpC, VpVp, 0);
  }

  // ----- attention softmaxes -----
  { dim3 g(Cn, Bn);
    softmax_bf16_kernel<<<g, 256, 0, stream>>>(e_t1, at1, Cn, Cn, Cn, 0, 0, CC, CC);
    softmax_bf16_kernel<<<g, 256, 0, stream>>>(e_t1, at2, Cn, Cn, Cn, 1, 0, CC, CC); }
  es_modulate_kernel<<<cdiv((long)Bn * Vn * Vn, EW), EW, 0, stream>>>(e_s1, sg, e_s1m);
  { dim3 g(Vn, Bn);
    softmax_bf16_kernel<<<g, 256, 0, stream>>>(e_s1m, as2T, Vn, Vp, Vp, 0, 1, VpVp, VpVp);
    softmax_bf16_kernel<<<g, 256, 0, stream>>>(e_s1m, as1T, Vn, Vp, Vp, 1, 1, VpVp, VpVp); }

  // ----- z1 = at2 @ g1 @ as2 ; z2 = at1 @ g2 @ as1 (stored transposed) -----
  {
    dim3 grid(Vp / 32, Cn / (32 * GEMM_WAVES), Bn);
    wmma_gemm_tn<false, true, false><<<grid, gblk, 0, stream>>>(
        at2, g1T, NOF, tmp1, nullptr, nullptr, Cn, Vp, Cn, Cn, Vn,
        Cn, Cn, Vp, 0, CC, VpC, CVp, 0);
    wmma_gemm_tn<false, false, true><<<grid, gblk, 0, stream>>>(
        tmp1, as2T, NOF, nullptr, z1T, nullptr, Cn, Vp, Vp, Cn, Vn,
        Vp, Vp, 0, Cn, CVp, VpVp, 0, VpC);
    wmma_gemm_tn<false, true, false><<<grid, gblk, 0, stream>>>(
        at1, g2T, NOF, tmp2, nullptr, nullptr, Cn, Vp, Cn, Cn, Vn,
        Cn, Cn, Vp, 0, CC, VpC, CVp, 0);
    wmma_gemm_tn<false, false, true><<<grid, gblk, 0, stream>>>(
        tmp2, as1T, NOF, nullptr, z2T, nullptr, Cn, Vp, Vp, Cn, Vn,
        Vp, Vp, 0, Cn, CVp, VpVp, 0, VpC);
  }

  // ----- conv = W_w @ z + W_b  -> (B,T,Vp) f32 -----
  {
    dim3 grid(Vp / 32, Tn / (32 * GEMM_WAVES), Bn);
    wmma_gemm_tn<true, false, false><<<grid, gblk, 0, stream>>>(
        Ww_bf, z1T, conv1f, nullptr, nullptr, W_b, Tn, Vp, Cn, Tn, Vn,
        Cn, Cn, Vp, 0, 0, VpC, TVp, 0);
    wmma_gemm_tn<true, false, false><<<grid, gblk, 0, stream>>>(
        Ww_bf, z2T, conv2f, nullptr, nullptr, W_b, Tn, Vp, Cn, Tn, Vn,
        Cn, Cn, Vp, 0, 0, VpC, TVp, 0);
  }

  // ----- output-side BN stats and fusion scalars -----
  bn_stats_kernel<<<Tn, 256, 0, stream>>>(conv1f, Vp, muc1, rsc1);
  bn_stats_kernel<<<Tn, 256, 0, stream>>>(conv2f, Vp, muc2, rsc2);
  mean_last_kernel<<<cdiv((long)Bn * Tn, EW), EW, 0, stream>>>(x1, x1_tm);
  mean_mid_kernel<<<cdiv((long)Bn * Vn, EW), EW, 0, stream>>>(x2, x2_sm);
  mean_all_kernel<<<Bn, 256, 0, stream>>>(x1, meanA1);
  mean_all_kernel<<<Bn, 256, 0, stream>>>(x2, meanA2);
  proj_mean_kernel<<<Bn, 256, 0, stream>>>(conv1f, muc1, rsc1, W_bn_g, W_bn_b, p1);
  proj_mean_kernel<<<Bn, 256, 0, stream>>>(conv2f, muc2, rsc2, W_bn_g, W_bn_b, p2);

  // ----- final fusion -----
  fuse_kernel<<<cdiv(BTV, EW), EW, 0, stream>>>(x1, x2_sm, 1, conv1f, muc1, rsc1,
                                                W_bn_g, W_bn_b, meanA1, meanA2, p1, out);
  fuse_kernel<<<cdiv(BTV, EW), EW, 0, stream>>>(x2, x1_tm, 0, conv2f, muc2, rsc2,
                                                W_bn_g, W_bn_b, meanA2, meanA1, p2, out + BTV);
}
